// FCMTLSTMFull_40561671143596
// MI455X (gfx1250) — matrix operand order repaired and checked by reading, not compiled, not run on hardware
//
#include <hip/hip_runtime.h>

// ---------------------------------------------------------------------------
// FCMTLSTMFull on MI455X (gfx1250): f16 WMMA pipeline, register-hoisted frags
//   B=8192, IN=184, H=128, SEQ=4, FPS=46
// ---------------------------------------------------------------------------

#define BATCH 8192
#define INF   184
#define HID   128
#define SEQN  4
#define FPS_  46
#define EPS_  1e-5f

typedef __attribute__((ext_vector_type(16))) _Float16 v16h;
typedef __attribute__((ext_vector_type(8)))  _Float16 v8h;
typedef __attribute__((ext_vector_type(8)))  float    v8f;

union HFrag { v16h v; v8h h[2]; };

__device__ inline v8f vzero8f() {
  v8f c = {0.f,0.f,0.f,0.f,0.f,0.f,0.f,0.f};
  return c;
}

// A fragment: 16x32 f16 tile, row-major source, lda in elements (mult of 8).
// lane L (<16): row L,      K = {k0.. k0+7} ∪ {k0+16..k0+23}
// lane L (>=16): row L-16,  K = {k0+8..k0+15} ∪ {k0+24..k0+31}
__device__ inline v16h load_a_frag(const _Float16* __restrict__ A, int lda, int lane) {
  int r = lane & 15, half = lane >> 4;
  const _Float16* p = A + r * lda + half * 8;
  HFrag f;
  f.h[0] = *(const v8h*)(p);
  f.h[1] = *(const v8h*)(p + 16);
  return f.v;
}

// B fragment: 32x16 f16 tile; source stored (N,K) row-major, ldb in elements.
// lane L: col L&15, K = 8*(L>>4) + {0..7} ∪ 16 + 8*(L>>4) + {0..7}
__device__ inline v16h load_b_frag(const _Float16* __restrict__ Bw, int ldb, int lane) {
  int c = lane & 15, half = lane >> 4;
  const _Float16* p = Bw + c * ldb + half * 8;
  HFrag f;
  f.h[0] = *(const v8h*)(p);
  f.h[1] = *(const v8h*)(p + 16);
  return f.v;
}

__device__ inline v8f wmma16(v16h a, v16h b, v8f c) {
  return __builtin_amdgcn_wmma_f32_16x16x32_f16(false, a, false, b, (short)0, c,
                                                false, false);
}

__device__ inline float sigmoidf_(float x) { return 1.f / (1.f + __expf(-x)); }

// ---------------------------------------------------------------------------
// Small prep kernels
// ---------------------------------------------------------------------------
__global__ __launch_bounds__(256) void k_cvt(_Float16* __restrict__ dst,
                                             const float* __restrict__ src, int n) {
  int i = blockIdx.x * 256 + threadIdx.x;
  if (i < n) dst[i] = (_Float16)src[i];
}

__global__ __launch_bounds__(256) void k_addf32(float* __restrict__ dst,
                                                const float* __restrict__ a,
                                                const float* __restrict__ b, int n) {
  int i = blockIdx.x * 256 + threadIdx.x;
  if (i < n) dst[i] = a[i] + b[i];
}

// conv2_w (256,128,3) -> w2r (256, 384) with K = toff*128 + ic
__global__ __launch_bounds__(256) void k_w2r(_Float16* __restrict__ dst,
                                             const float* __restrict__ src) {
  int i = blockIdx.x * 256 + threadIdx.x;
  if (i >= 256 * 384) return;
  int oc = i / 384, rem = i % 384;
  int toff = rem >> 7, ic = rem & 127;
  dst[i] = (_Float16)src[oc * 384 + ic * 3 + toff];
}

// Wih0 (2,512,46) -> (2,512,64) zero-padded
__global__ __launch_bounds__(256) void k_wih0p(_Float16* __restrict__ dst,
                                               const float* __restrict__ src) {
  int i = blockIdx.x * 256 + threadIdx.x;
  if (i >= 2 * 512 * 64) return;
  int kk = i & 63, row = i >> 6; // row = d*512+n
  dst[i] = (kk < FPS_) ? (_Float16)src[row * FPS_ + kk] : (_Float16)0.f;
}

// xt_pad[s][b][jj] = (jj<46) ? x[b,:] . fp_w[46s+jj,:] + fp_b : 0   (f16)
__global__ __launch_bounds__(256) void k_xt(const float* __restrict__ x,
                                            const float* __restrict__ fp_w,
                                            const float* __restrict__ fp_b,
                                            _Float16* __restrict__ xt) {
  int gid = blockIdx.x * 256 + threadIdx.x;   // BATCH*4*64 total
  int b = gid >> 8;
  int rem = gid & 255;
  int s = rem >> 6, jj = rem & 63;
  float v = 0.f;
  if (jj < FPS_) {
    int j = s * FPS_ + jj;
    const float* xr = x + (size_t)b * INF;
    const float* wr = fp_w + (size_t)j * INF;
    float acc = fp_b[j];
    for (int i = 0; i < INF; ++i) acc += xr[i] * wr[i];
    v = acc;
  }
  xt[((size_t)s * BATCH + b) * 64 + jj] = (_Float16)v;
}

// ---------------------------------------------------------------------------
// Fused conv1+bn+relu -> LDS, conv2 (WMMA im2col GEMM) + max + bn2 + relu
// One workgroup (256 thr / 8 waves) per sample. B-fragments (weights) are
// hoisted into registers (12 x v16h = 96 VGPRs) and reused across all 12
// M-tiles; A streams from LDS.
// ---------------------------------------------------------------------------
#define CROWS 194   // 1 zero pad row + 184 + 9 pad (M padded to 192 rows)
__global__ __launch_bounds__(256) void k_conv(
    const float* __restrict__ x, const float* __restrict__ w1,
    const float* __restrict__ b1, const float* __restrict__ g1,
    const float* __restrict__ be1, const _Float16* __restrict__ w2r,
    const float* __restrict__ b2, const float* __restrict__ g2,
    const float* __restrict__ be2, _Float16* __restrict__ hmax) {
  __shared__ __align__(16) _Float16 h1[CROWS * 128];
  __shared__ float xs[INF];
  __shared__ float w1s[384];
  __shared__ float sc1[128], sh1[128];

  int b = blockIdx.x, tid = threadIdx.x;
  if (tid < INF) xs[tid] = x[(size_t)b * INF + tid];
  for (int i = tid; i < 384; i += 256) w1s[i] = w1[i];
  if (tid < 128) {
    float s = g1[tid] * rsqrtf(1.f + EPS_);
    sc1[tid] = s;
    sh1[tid] = b1[tid] * s + be1[tid];
  }
  for (int i = tid; i < 128; i += 256) h1[i] = (_Float16)0.f;                 // pad row 0
  for (int i = tid; i < 9 * 128; i += 256) h1[185 * 128 + i] = (_Float16)0.f; // rows 185..193
  __syncthreads();

  // conv1 + bn1 + relu -> h1 rows 1..184
  for (int idx = tid; idx < INF * 128; idx += 256) {
    int t = idx >> 7, oc = idx & 127;
    float acc = xs[t] * w1s[oc * 3 + 1];
    if (t > 0)       acc += xs[t - 1] * w1s[oc * 3 + 0];
    if (t < INF - 1) acc += xs[t + 1] * w1s[oc * 3 + 2];
    acc = acc * sc1[oc] + sh1[oc];
    h1[(t + 1) * 128 + oc] = (_Float16)fmaxf(acc, 0.f);
  }
  __syncthreads();

  // conv2 as GEMM: A row t = &h1[t*128] (K=384 contiguous), B = w2r (N,K)
  int wave = tid >> 5, lane = tid & 31;
  int hi = lane >> 4;
  float mx[2];
  for (int nt = 0; nt < 2; ++nt) {
    int n0 = (wave * 2 + nt) * 16;
    // hoist all 12 K-step weight fragments for this N-tile
    v16h bfr[12];
#pragma unroll
    for (int kt = 0; kt < 12; ++kt)
      bfr[kt] = load_b_frag(w2r + (size_t)n0 * 384 + kt * 32, 384, lane);
    float mxv = -3.0e38f;
    for (int mt = 0; mt < 12; ++mt) {
      const _Float16* Ab = &h1[mt * 16 * 128];
      v8f c = vzero8f();
#pragma unroll
      for (int kt = 0; kt < 12; ++kt) {
        v16h a = load_a_frag(Ab + kt * 32, 128, lane);
        c = wmma16(a, bfr[kt], c);
      }
      float m = -3.0e38f;
      int rbase = mt * 16 + hi * 8;
#pragma unroll
      for (int r = 0; r < 8; ++r)
        if (rbase + r < INF) m = fmaxf(m, c[r]);
      m = fmaxf(m, __shfl_xor(m, 16, 32));
      mxv = fmaxf(mxv, m);
    }
    mx[nt] = mxv;
  }
  if (lane < 16) {
#pragma unroll
    for (int nt = 0; nt < 2; ++nt) {
      int n = (wave * 2 + nt) * 16 + lane;
      float y = (mx[nt] + b2[n]) * (g2[n] * rsqrtf(1.f + EPS_)) + be2[n];
      hmax[(size_t)b * 256 + n] = (_Float16)fmaxf(y, 0.f);
    }
  }
}

// ---------------------------------------------------------------------------
// Generic WMMA GEMM: out[M=8192, N] = epi(A[M,K] @ Bw[N,K]^T + bias)
// grid (M/128, N/128), 256 thr (wave = M-tile, 8 N-tiles per wave).
// A fragments hoisted into registers (K/32 x v16h), reused across 8 N-tiles.
// EPI 0: +bias ; EPI 1: +bias, BN(g,bet), ReLU. out ld = ldo, f16.
// ---------------------------------------------------------------------------
template <int K, int EPI>
__global__ __launch_bounds__(256) void k_gemm(
    const _Float16* __restrict__ A, const _Float16* __restrict__ Bw,
    const float* __restrict__ bias, const float* __restrict__ gam,
    const float* __restrict__ bet, _Float16* __restrict__ out, int ldo) {
  int wave = threadIdx.x >> 5, lane = threadIdx.x & 31;
  int m0 = blockIdx.x * 128 + wave * 16;
  int nblk = blockIdx.y * 128;
  int col = lane & 15, hi = lane >> 4;

  v16h afr[K / 32];
#pragma unroll
  for (int kt = 0; kt < K / 32; ++kt)
    afr[kt] = load_a_frag(A + (size_t)m0 * K + kt * 32, K, lane);

  for (int j = 0; j < 8; ++j) {
    int n0 = nblk + j * 16;
    v8f c = vzero8f();
#pragma unroll
    for (int kt = 0; kt < K / 32; ++kt) {
      v16h bf = load_b_frag(Bw + (size_t)n0 * K + kt * 32, K, lane);
      c = wmma16(afr[kt], bf, c);
    }
    int n = n0 + col;
    float bs = bias[n];
    float sc = 1.f, sh = 0.f;
    if (EPI == 1) { sc = gam[n] * rsqrtf(1.f + EPS_); sh = bet[n]; }
#pragma unroll
    for (int r = 0; r < 8; ++r) {
      float y = c[r] + bs;
      if (EPI == 1) y = fmaxf(y * sc + sh, 0.f);
      out[(size_t)(m0 + r + 8 * hi) * ldo + n] = (_Float16)y;
    }
  }
}

// ---------------------------------------------------------------------------
// BiLSTM layer: 1024 threads (32 waves), WG = 64 batch rows x 1 direction.
// wave = (mtile in 0..3, htile in 0..7); c state lives in C-frag layout regs.
// x-input and h-recurrent A fragments hoisted once per timestep, reused by
// all four gates.
// xin (SEQ, B, KIN) f16 ; Wih (2,512,KIN) f16 ; Whh (2,512,128) f16 ;
// bias (2,512) f32 ; hout (SEQ, B, 256) f16 (dir selects column half)
// ---------------------------------------------------------------------------
template <int KIN>
__global__ __launch_bounds__(1024) void k_lstm(
    const _Float16* __restrict__ xin, const _Float16* __restrict__ Wih,
    const _Float16* __restrict__ Whh, const float* __restrict__ bias,
    _Float16* __restrict__ hout) {
  __shared__ __align__(16) _Float16 hsh[64 * 128];
  int wg = blockIdx.x;
  int dir = wg & 1, b0 = (wg >> 1) * 64;
  int tid = threadIdx.x, wave = tid >> 5, lane = tid & 31;
  int mt = wave & 3, ht = wave >> 2;
  int col = lane & 15, hi = lane >> 4;
  const _Float16* WihD = Wih + (size_t)dir * 512 * KIN;
  const _Float16* WhhD = Whh + (size_t)dir * 512 * 128;
  const float* biasD = bias + dir * 512;

  for (int i = tid; i < 64 * 128; i += 1024) hsh[i] = (_Float16)0.f;
  v8f cstate = vzero8f();
  __syncthreads();

  for (int step = 0; step < SEQN; ++step) {
    int t = dir ? (SEQN - 1 - step) : step;
    const _Float16* xrow = xin + ((size_t)t * BATCH + b0 + mt * 16) * KIN;

    // hoist A fragments (shared by all four gates)
    v16h ax[KIN / 32];
#pragma unroll
    for (int kt = 0; kt < KIN / 32; ++kt)
      ax[kt] = load_a_frag(xrow + kt * 32, KIN, lane);
    v16h ah[4];
#pragma unroll
    for (int kt = 0; kt < 4; ++kt)
      ah[kt] = load_a_frag(&hsh[mt * 16 * 128 + kt * 32], 128, lane);

    v8f acc[4];
#pragma unroll
    for (int gi = 0; gi < 4; ++gi) {
      acc[gi] = vzero8f();
      int n0 = gi * 128 + ht * 16;
#pragma unroll
      for (int kt = 0; kt < KIN / 32; ++kt) {
        v16h bf = load_b_frag(WihD + (size_t)n0 * KIN + kt * 32, KIN, lane);
        acc[gi] = wmma16(ax[kt], bf, acc[gi]);
      }
#pragma unroll
      for (int kt = 0; kt < 4; ++kt) {
        v16h bf = load_b_frag(WhhD + (size_t)n0 * 128 + kt * 32, 128, lane);
        acc[gi] = wmma16(ah[kt], bf, acc[gi]);
      }
      float bn = biasD[n0 + col];
#pragma unroll
      for (int r = 0; r < 8; ++r) acc[gi][r] += bn;
    }
    __syncthreads();  // all waves done reading hsh for this step

    float hnew[8];
#pragma unroll
    for (int r = 0; r < 8; ++r) {
      float ig = sigmoidf_(acc[0][r]);
      float fg = sigmoidf_(acc[1][r]);
      float gg = tanhf(acc[2][r]);
      float og = sigmoidf_(acc[3][r]);
      float cn = fg * cstate[r] + ig * gg;
      cstate[r] = cn;
      hnew[r] = og * tanhf(cn);
    }
    int hcol = ht * 16 + col;
    _Float16* hg = hout + ((size_t)t * BATCH + b0 + mt * 16) * 256 + dir * 128;
#pragma unroll
    for (int r = 0; r < 8; ++r) {
      int row = r + 8 * hi;
      _Float16 hv = (_Float16)hnew[r];
      hsh[(mt * 16 + row) * 128 + hcol] = hv;
      hg[(size_t)row * 256 + hcol] = hv;
    }
    __syncthreads();  // hsh ready for next step
  }
}

// ---------------------------------------------------------------------------
// Attention scores: score[m] = sum_n tanh( (h2 @ attW^T)[m,n] + attb[n] )*attv[n]
// m = s*B + b.  WMMA GEMM fused with tanh + attv dot + lane butterfly.
// A fragments hoisted, reused across all 16 N-tiles.
// ---------------------------------------------------------------------------
__global__ __launch_bounds__(256) void k_att1(const _Float16* __restrict__ h2,
                                              const _Float16* __restrict__ attW,
                                              const float* __restrict__ attb,
                                              const float* __restrict__ attv,
                                              float* __restrict__ scores) {
  int wave = threadIdx.x >> 5, lane = threadIdx.x & 31;
  int m0 = blockIdx.x * 128 + wave * 16;
  int col = lane & 15, hi = lane >> 4;

  v16h afr[8];
#pragma unroll
  for (int kt = 0; kt < 8; ++kt)
    afr[kt] = load_a_frag(h2 + (size_t)m0 * 256 + kt * 32, 256, lane);

  float sacc[8] = {0.f, 0.f, 0.f, 0.f, 0.f, 0.f, 0.f, 0.f};
  for (int nt = 0; nt < 16; ++nt) {
    int n0 = nt * 16;
    v8f c = vzero8f();
#pragma unroll
    for (int kt = 0; kt < 8; ++kt) {
      v16h bf = load_b_frag(attW + (size_t)n0 * 256 + kt * 32, 256, lane);
      c = wmma16(afr[kt], bf, c);
    }
    float bn = attb[n0 + col];
    float av = attv[n0 + col];
#pragma unroll
    for (int r = 0; r < 8; ++r) sacc[r] += tanhf(c[r] + bn) * av;
  }
#pragma unroll
  for (int m = 1; m < 16; m <<= 1)
#pragma unroll
    for (int r = 0; r < 8; ++r) sacc[r] += __shfl_xor(sacc[r], m, 32);
  if (col == 0) {
#pragma unroll
    for (int r = 0; r < 8; ++r) scores[m0 + r + 8 * hi] = sacc[r];
  }
}

// softmax over SEQ + context -> z[:, 256:512]. one thread = (b, 8 channels)
__global__ __launch_bounds__(256) void k_att2(const float* __restrict__ scores,
                                              const _Float16* __restrict__ h2,
                                              _Float16* __restrict__ z) {
  int gid = blockIdx.x * 256 + threadIdx.x;  // BATCH*32
  int b = gid >> 5;
  int ch0 = (gid & 31) * 8;
  float s[SEQN], mx = -3.0e38f;
#pragma unroll
  for (int i = 0; i < SEQN; ++i) { s[i] = scores[i * BATCH + b]; mx = fmaxf(mx, s[i]); }
  float e[SEQN], sum = 0.f;
#pragma unroll
  for (int i = 0; i < SEQN; ++i) { e[i] = __expf(s[i] - mx); sum += e[i]; }
  float inv = 1.f / sum;
  float ctx[8] = {0.f, 0.f, 0.f, 0.f, 0.f, 0.f, 0.f, 0.f};
#pragma unroll
  for (int i = 0; i < SEQN; ++i) {
    float aw = e[i] * inv;
    const _Float16* hr = h2 + ((size_t)i * BATCH + b) * 256 + ch0;
#pragma unroll
    for (int j = 0; j < 8; ++j) ctx[j] += aw * (float)hr[j];
  }
#pragma unroll
  for (int j = 0; j < 8; ++j) z[(size_t)b * 512 + 256 + ch0 + j] = (_Float16)ctx[j];
}

// ---------------------------------------------------------------------------
// Decoders: wave per sample; only the group-selected decoder is computed.
// ---------------------------------------------------------------------------
__global__ __launch_bounds__(256) void k_dec(
    const _Float16* __restrict__ enc, const int* __restrict__ labels,
    const float* __restrict__ sw1, const float* __restrict__ sb1,
    const float* __restrict__ sw2, const float* __restrict__ sb2,
    const float* __restrict__ lw1, const float* __restrict__ lb1,
    const float* __restrict__ lw2, const float* __restrict__ lb2,
    float* __restrict__ out) {
  __shared__ float er[8][256];
  int wave = threadIdx.x >> 5, lane = threadIdx.x & 31;
  int b = blockIdx.x * 8 + wave;
  for (int j = lane; j < 256; j += 32) er[wave][j] = (float)enc[(size_t)b * 256 + j];
  __syncthreads();
  int g = labels[b];
  const float *w1, *b1, *w2;
  float b2v;
  int H1;
  if (g < 2) {
    w1 = sw1 + (size_t)g * 128 * 256; b1 = sb1 + g * 128;
    w2 = sw2 + g * 128; b2v = sb2[g]; H1 = 128;
  } else {
    int q = g - 2;
    w1 = lw1 + (size_t)q * 256 * 256; b1 = lb1 + q * 256;
    w2 = lw2 + q * 256; b2v = lb2[q]; H1 = 256;
  }
  float acc = 0.f;
  for (int jj = lane; jj < H1; jj += 32) {
    const float* wr = w1 + (size_t)jj * 256;
    float d = b1[jj];
    for (int i = 0; i < 256; ++i) d += er[wave][i] * wr[i];
    acc += fmaxf(d, 0.f) * w2[jj];
  }
#pragma unroll
  for (int m = 1; m < 32; m <<= 1) acc += __shfl_xor(acc, m, 32);
  if (lane == 0) out[b] = acc + b2v;
}

// ---------------------------------------------------------------------------
// Host launcher
// ---------------------------------------------------------------------------
extern "C" void kernel_launch(void* const* d_in, const int* in_sizes, int n_in,
                              void* d_out, int out_size, void* d_ws, size_t ws_size,
                              hipStream_t stream) {
  (void)in_sizes; (void)n_in; (void)out_size; (void)ws_size;
  const float* x        = (const float*)d_in[0];
  const int*   labels   = (const int*)d_in[1];
  const float* conv1_w  = (const float*)d_in[2];
  const float* conv1_b  = (const float*)d_in[3];
  const float* bn1_g    = (const float*)d_in[4];
  const float* bn1_b    = (const float*)d_in[5];
  const float* conv2_w  = (const float*)d_in[6];
  const float* conv2_b  = (const float*)d_in[7];
  const float* bn2_g    = (const float*)d_in[8];
  const float* bn2_b    = (const float*)d_in[9];
  const float* cnn_dw   = (const float*)d_in[10];
  const float* cnn_db   = (const float*)d_in[11];
  const float* fp_w     = (const float*)d_in[12];
  const float* fp_b     = (const float*)d_in[13];
  const float* Wih0     = (const float*)d_in[14];
  const float* Whh0     = (const float*)d_in[15];
  const float* bih0     = (const float*)d_in[16];
  const float* bhh0     = (const float*)d_in[17];
  const float* Wih1     = (const float*)d_in[18];
  const float* Whh1     = (const float*)d_in[19];
  const float* bih1     = (const float*)d_in[20];
  const float* bhh1     = (const float*)d_in[21];
  const float* attW_w   = (const float*)d_in[22];
  const float* attW_b   = (const float*)d_in[23];
  const float* attv     = (const float*)d_in[24];
  const float* enc1_w   = (const float*)d_in[25];
  const float* enc1_b   = (const float*)d_in[26];
  const float* bne1_g   = (const float*)d_in[27];
  const float* bne1_b   = (const float*)d_in[28];
  const float* enc2_w   = (const float*)d_in[29];
  const float* enc2_b   = (const float*)d_in[30];
  const float* bne2_g   = (const float*)d_in[31];
  const float* bne2_b   = (const float*)d_in[32];
  const float* decS_w1  = (const float*)d_in[33];
  const float* decS_b1  = (const float*)d_in[34];
  const float* decS_w2  = (const float*)d_in[35];
  const float* decS_b2  = (const float*)d_in[36];
  const float* decL_w1  = (const float*)d_in[37];
  const float* decL_b1  = (const float*)d_in[38];
  const float* decL_w2  = (const float*)d_in[39];
  const float* decL_b2  = (const float*)d_in[40];
  float* preds = (float*)d_out;

  // bump allocator on workspace (256B aligned)
  char* p = (char*)d_ws;
  auto alloc = [&](size_t bytes) -> char* {
    char* r = p;
    p += (bytes + 255) & ~(size_t)255;
    return r;
  };
  _Float16* w2r     = (_Float16*)alloc(256 * 384 * 2);
  _Float16* wih0p   = (_Float16*)alloc(2 * 512 * 64 * 2);
  _Float16* whh0h   = (_Float16*)alloc(2 * 512 * 128 * 2);
  _Float16* wih1h   = (_Float16*)alloc(2 * 512 * 256 * 2);
  _Float16* whh1h   = (_Float16*)alloc(2 * 512 * 128 * 2);
  _Float16* cnn_dwh = (_Float16*)alloc(256 * 256 * 2);
  _Float16* attWh   = (_Float16*)alloc(256 * 256 * 2);
  _Float16* enc1wh  = (_Float16*)alloc(512 * 512 * 2);
  _Float16* enc2wh  = (_Float16*)alloc(256 * 512 * 2);
  float*    bias0   = (float*)alloc(2 * 512 * 4);
  float*    bias1   = (float*)alloc(2 * 512 * 4);
  _Float16* xt_pad  = (_Float16*)alloc((size_t)SEQN * BATCH * 64 * 2);
  _Float16* hmax    = (_Float16*)alloc((size_t)BATCH * 256 * 2);
  _Float16* h1cat   = (_Float16*)alloc((size_t)SEQN * BATCH * 256 * 2);
  _Float16* h2cat   = (_Float16*)alloc((size_t)SEQN * BATCH * 256 * 2);
  float*    scores  = (float*)alloc((size_t)SEQN * BATCH * 4);
  _Float16* zbuf    = (_Float16*)alloc((size_t)BATCH * 512 * 2);
  _Float16* zz      = (_Float16*)alloc((size_t)BATCH * 512 * 2);
  _Float16* encb    = (_Float16*)alloc((size_t)BATCH * 256 * 2);

  auto grid1 = [](int n) { return dim3((n + 255) / 256); };

  // ---- weight repack / conversion ----
  k_cvt<<<grid1(2 * 512 * 128), 256, 0, stream>>>(whh0h, Whh0, 2 * 512 * 128);
  k_cvt<<<grid1(2 * 512 * 256), 256, 0, stream>>>(wih1h, Wih1, 2 * 512 * 256);
  k_cvt<<<grid1(2 * 512 * 128), 256, 0, stream>>>(whh1h, Whh1, 2 * 512 * 128);
  k_cvt<<<grid1(256 * 256),     256, 0, stream>>>(cnn_dwh, cnn_dw, 256 * 256);
  k_cvt<<<grid1(256 * 256),     256, 0, stream>>>(attWh, attW_w, 256 * 256);
  k_cvt<<<grid1(512 * 512),     256, 0, stream>>>(enc1wh, enc1_w, 512 * 512);
  k_cvt<<<grid1(256 * 512),     256, 0, stream>>>(enc2wh, enc2_w, 256 * 512);
  k_w2r<<<grid1(256 * 384), 256, 0, stream>>>(w2r, conv2_w);
  k_wih0p<<<grid1(2 * 512 * 64), 256, 0, stream>>>(wih0p, Wih0);
  k_addf32<<<grid1(1024), 256, 0, stream>>>(bias0, bih0, bhh0, 1024);
  k_addf32<<<grid1(1024), 256, 0, stream>>>(bias1, bih1, bhh1, 1024);

  // ---- fp projection -> xt_pad (SEQ, B, 64) ----
  k_xt<<<grid1(BATCH * SEQN * 64), 256, 0, stream>>>(x, fp_w, fp_b, xt_pad);

  // ---- fused conv stack -> hmax (B, 256) ----
  k_conv<<<dim3(BATCH), 256, 0, stream>>>(x, conv1_w, conv1_b, bn1_g, bn1_b, w2r,
                                          conv2_b, bn2_g, bn2_b, hmax);

  // ---- spatial = hmax @ cnn_dw^T + cnn_db  -> z[:, 0:256] ----
  k_gemm<256, 0><<<dim3(BATCH / 128, 2), 256, 0, stream>>>(
      hmax, cnn_dwh, cnn_db, nullptr, nullptr, zbuf, 512);

  // ---- BiLSTM layers ----
  k_lstm<64><<<dim3((BATCH / 64) * 2), 1024, 0, stream>>>(xt_pad, wih0p, whh0h,
                                                          bias0, h1cat);
  k_lstm<256><<<dim3((BATCH / 64) * 2), 1024, 0, stream>>>(h1cat, wih1h, whh1h,
                                                           bias1, h2cat);

  // ---- attention ----
  k_att1<<<dim3(SEQN * BATCH / 128), 256, 0, stream>>>(h2cat, attWh, attW_b, attv,
                                                       scores);
  k_att2<<<grid1(BATCH * 32), 256, 0, stream>>>(scores, h2cat, zbuf);

  // ---- encoder ----
  k_gemm<512, 1><<<dim3(BATCH / 128, 4), 256, 0, stream>>>(
      zbuf, enc1wh, enc1_b, bne1_g, bne1_b, zz, 512);
  k_gemm<512, 1><<<dim3(BATCH / 128, 2), 256, 0, stream>>>(
      zz, enc2wh, enc2_b, bne2_g, bne2_b, encb, 256);

  // ---- group-selected decoders -> preds (B,1) ----
  k_dec<<<dim3(BATCH / 8), 256, 0, stream>>>(encb, labels, decS_w1, decS_b1,
                                             decS_w2, decS_b2, decL_w1, decL_b1,
                                             decL_w2, decL_b2, preds);
}
